// MaskFusionThorMoE_7533372638012
// MI455X (gfx1250) — compile-verified, matrix operand order checked by reading, not run on hardware
//
#include <hip/hip_runtime.h>
#include <hip/hip_bf16.h>
#include <math.h>

#define NE 8
#define NT 8192
#define NH 768
#define NI 3072
#define LN_EPS 1e-12f

typedef __bf16 bf16_t;
typedef __attribute__((ext_vector_type(16))) __bf16 v16bf;
typedef __attribute__((ext_vector_type(8)))  __bf16 v8bf;
typedef __attribute__((ext_vector_type(8)))  float  v8f;

union FragBF {
  v16bf v;
  v8bf  h[2];
};

// ---------------- CDNA5 async global->LDS staging ----------------
__device__ __forceinline__ void async_b128(unsigned int lds_off, const bf16_t* g) {
  // GLOBAL_LOAD_ASYNC_TO_LDS_B128 (GV mode): VDST = LDS byte offset VGPR,
  // VADDR = 64-bit global address. Tracked by ASYNCcnt.
  asm volatile("global_load_async_to_lds_b128 %0, %1, off"
               :: "v"(lds_off), "v"(g) : "memory");
}
__device__ __forceinline__ void wait_async0() {
  asm volatile("s_wait_asynccnt 0" ::: "memory");
}
// Async loads retire in order: waiting for <=4 outstanding guarantees the
// previous buffer's 4 loads are done while the 4 just-issued stay in flight.
__device__ __forceinline__ void wait_async4() {
  asm volatile("s_wait_asynccnt 4" ::: "memory");
}
__device__ __forceinline__ void stage4(unsigned int lds_a, unsigned int lds_b,
                                       const bf16_t* a, const bf16_t* b) {
  async_b128(lds_a,      a);
  async_b128(lds_a + 16, a + 8);
  async_b128(lds_b,      b);
  async_b128(lds_b + 16, b + 8);
}

// ---------------- conversion (linear) ----------------
__global__ void k_f32_to_bf16(const float* __restrict__ in,
                              bf16_t* __restrict__ out, size_t n) {
  size_t i = blockIdx.x * (size_t)blockDim.x + threadIdx.x;
  size_t stride = (size_t)gridDim.x * blockDim.x;
  for (; i < n; i += stride) out[i] = (bf16_t)in[i];
}

// ---------------- conversion + transpose: in[e][K][N] f32 -> out[e][N][K] bf16 ----
__global__ __launch_bounds__(256) void k_transpose_bf16(
    const float* __restrict__ in, bf16_t* __restrict__ out, int K, int N) {
  __shared__ float tile[32][33];
  const int e = blockIdx.z;
  const float* src = in + (size_t)e * K * N;
  bf16_t* dst = out + (size_t)e * K * N;
  const int k0 = blockIdx.y * 32;
  const int n0 = blockIdx.x * 32;
  const int tx = threadIdx.x & 31;
  const int ty = threadIdx.x >> 5;   // 0..7
#pragma unroll
  for (int i = 0; i < 32; i += 8)
    tile[ty + i][tx] = src[(size_t)(k0 + ty + i) * N + (n0 + tx)];
  __syncthreads();
#pragma unroll
  for (int i = 0; i < 32; i += 8)
    dst[(size_t)(n0 + ty + i) * K + (k0 + tx)] = (bf16_t)tile[tx][ty + i];
}

// ---------------- routing ----------------
__global__ void k_route_init(int* counts, int* cursor) {
  if (threadIdx.x < NE) { counts[threadIdx.x] = 0; cursor[threadIdx.x] = 0; }
}
__global__ void k_route_count(const int* __restrict__ eid, int* counts) {
  int t = blockIdx.x * blockDim.x + threadIdx.x;
  if (t < NT) atomicAdd(&counts[eid[t]], 1);
}
__global__ void k_route_scan(const int* __restrict__ counts, int* offsets) {
  if (threadIdx.x == 0) {
    int s = 0;
    for (int e = 0; e < NE; ++e) { offsets[e] = s; s += counts[e]; }
  }
}
__global__ void k_route_fill(const int* __restrict__ eid,
                             const int* __restrict__ offsets,
                             int* cursor, int* __restrict__ sorted) {
  int t = blockIdx.x * blockDim.x + threadIdx.x;
  if (t < NT) {
    int e = eid[t];
    int p = atomicAdd(&cursor[e], 1);
    sorted[offsets[e] + p] = t;
  }
}

// ---------------- WMMA fragment loader (ISA 7.12.2 16-bit layout) ----------
// lane<16: row/col = lane&15, K = {kb..kb+7, kb+16..kb+23} with kb=0;
// lane>=16: kb=8. Two contiguous 16B LDS reads.
__device__ __forceinline__ v16bf load_frag16(const bf16_t* S, int row_base,
                                             int lane, int ldk) {
  int row = row_base + (lane & 15);
  int kb  = (lane >> 4) << 3;
  const bf16_t* p = S + row * ldk + kb;
  FragBF f;
  f.h[0] = *(const v8bf*)(p);
  f.h[1] = *(const v8bf*)(p + 16);
  return f.v;
}

#define BM 128
#define BN 128
#define BK 32
#define LDK 40    // BK + 8 pad; 80B row keeps 16B alignment, kills bank conflicts

// one 32-deep K-slab of WMMAs for this wave's 64x32 sub-tile
__device__ __forceinline__ void mma_tile(const bf16_t* As, const bf16_t* Bs,
                                         int waveM, int waveN, int lane,
                                         v8f acc[4][2]) {
  v16bf bfrag[2];
#pragma unroll
  for (int ni = 0; ni < 2; ++ni)
    bfrag[ni] = load_frag16(Bs, waveN * 32 + ni * 16, lane, LDK);
#pragma unroll
  for (int mi = 0; mi < 4; ++mi) {
    v16bf afrag = load_frag16(As, waveM * 64 + mi * 16, lane, LDK);
#pragma unroll
    for (int ni = 0; ni < 2; ++ni)
      acc[mi][ni] = __builtin_amdgcn_wmma_f32_16x16x32_bf16(
          false, afrag, false, bfrag[ni], (short)0, acc[mi][ni], false, false);
  }
}

// ---------------- grouped GEMM1: inter = gelu(Xg @ W1 + b1) ----------------
// w1t layout: [E][NI][NH]  (N-major, K contiguous)
__global__ __launch_bounds__(256) void k_ffn1(
    const bf16_t* __restrict__ xb, const bf16_t* __restrict__ w1t,
    const float* __restrict__ b1, const int* __restrict__ counts,
    const int* __restrict__ offsets, const int* __restrict__ sorted,
    bf16_t* __restrict__ inter) {
  __shared__ __attribute__((aligned(16))) bf16_t As0[BM * LDK];
  __shared__ __attribute__((aligned(16))) bf16_t As1[BM * LDK];
  __shared__ __attribute__((aligned(16))) bf16_t Bs0[BN * LDK];
  __shared__ __attribute__((aligned(16))) bf16_t Bs1[BN * LDK];

  const int e   = blockIdx.y >> 6;   // 64 M-tiles max per expert
  const int mt  = blockIdx.y & 63;
  const int cnt = counts[e];
  if (mt * BM >= cnt) return;
  const int off = offsets[e];
  const int n0  = blockIdx.x * BN;

  const int tid   = threadIdx.x;
  const int lane  = tid & 31;
  const int wid   = tid >> 5;
  const int waveM = wid >> 2;        // 0..1 -> 64 rows each
  const int waveN = wid & 3;         // 0..3 -> 32 cols each

  const int row = tid >> 1;          // 0..127
  const int seg = tid & 1;           // 16-element half of the 32-wide K slab
  const int gr  = mt * BM + row;
  const int tok = sorted[off + ((gr < cnt) ? gr : 0)];
  const bf16_t* a_src = xb + (size_t)tok * NH + seg * 16;
  const bf16_t* b_src = w1t + ((size_t)e * NI + n0 + row) * NH + seg * 16;

  const unsigned int lds_a0 = (unsigned int)(uintptr_t)&As0[row * LDK + seg * 16];
  const unsigned int lds_a1 = (unsigned int)(uintptr_t)&As1[row * LDK + seg * 16];
  const unsigned int lds_b0 = (unsigned int)(uintptr_t)&Bs0[row * LDK + seg * 16];
  const unsigned int lds_b1 = (unsigned int)(uintptr_t)&Bs1[row * LDK + seg * 16];

  v8f zero = {};
  v8f acc[4][2];
#pragma unroll
  for (int i = 0; i < 4; ++i)
#pragma unroll
    for (int j = 0; j < 2; ++j) acc[i][j] = zero;

  // software pipeline: buffer0 primed; each stage issues the next slab before
  // waiting on the current one (s_wait_asynccnt 4 keeps next-in-flight).
  stage4(lds_a0, lds_b0, a_src, b_src);
  for (int k0 = 0; k0 < NH; k0 += 2 * BK) {
    stage4(lds_a1, lds_b1, a_src + k0 + BK, b_src + k0 + BK);
    wait_async4();
    __syncthreads();
    mma_tile(As0, Bs0, waveM, waveN, lane, acc);
    __syncthreads();
    if (k0 + 2 * BK < NH) {
      stage4(lds_a0, lds_b0, a_src + k0 + 2 * BK, b_src + k0 + 2 * BK);
      wait_async4();
    } else {
      wait_async0();
    }
    __syncthreads();
    mma_tile(As1, Bs1, waveM, waveN, lane, acc);
    __syncthreads();
  }

  // epilogue: + bias, exact GELU, store bf16 in sorted order
  const int ln = lane & 15;
  const int lh = (lane >> 4) << 3;   // C/D: M = j + 8*(lane>=16), N = lane&15
#pragma unroll
  for (int mi = 0; mi < 4; ++mi) {
#pragma unroll
    for (int ni = 0; ni < 2; ++ni) {
      int n = n0 + waveN * 32 + ni * 16 + ln;
      float bias = b1[e * NI + n];
#pragma unroll
      for (int j = 0; j < 8; ++j) {
        int m = mt * BM + waveM * 64 + mi * 16 + lh + j;
        if (m < cnt) {
          float v = acc[mi][ni][j] + bias;
          v = 0.5f * v * (1.0f + erff(v * 0.70710678118654752f));
          inter[(size_t)(off + m) * NI + n] = (bf16_t)v;
        }
      }
    }
  }
}

// ---------------- grouped GEMM2: y = inter @ W2 + b2, scatter to token ----
// w2t layout: [E][NH][NI]  (N-major, K contiguous)
__global__ __launch_bounds__(256) void k_ffn2(
    const bf16_t* __restrict__ inter, const bf16_t* __restrict__ w2t,
    const float* __restrict__ b2, const int* __restrict__ counts,
    const int* __restrict__ offsets, const int* __restrict__ sorted,
    float* __restrict__ yffn) {
  __shared__ __attribute__((aligned(16))) bf16_t As0[BM * LDK];
  __shared__ __attribute__((aligned(16))) bf16_t As1[BM * LDK];
  __shared__ __attribute__((aligned(16))) bf16_t Bs0[BN * LDK];
  __shared__ __attribute__((aligned(16))) bf16_t Bs1[BN * LDK];

  const int e   = blockIdx.y >> 6;
  const int mt  = blockIdx.y & 63;
  const int cnt = counts[e];
  if (mt * BM >= cnt) return;
  const int off = offsets[e];
  const int n0  = blockIdx.x * BN;

  const int tid   = threadIdx.x;
  const int lane  = tid & 31;
  const int wid   = tid >> 5;
  const int waveM = wid >> 2;
  const int waveN = wid & 3;

  const int row = tid >> 1;
  const int seg = tid & 1;
  const int gr  = mt * BM + row;
  const int prow = off + ((gr < cnt) ? gr : 0);
  const bf16_t* a_src = inter + (size_t)prow * NI + seg * 16;
  const bf16_t* b_src = w2t + ((size_t)e * NH + n0 + row) * NI + seg * 16;

  const unsigned int lds_a0 = (unsigned int)(uintptr_t)&As0[row * LDK + seg * 16];
  const unsigned int lds_a1 = (unsigned int)(uintptr_t)&As1[row * LDK + seg * 16];
  const unsigned int lds_b0 = (unsigned int)(uintptr_t)&Bs0[row * LDK + seg * 16];
  const unsigned int lds_b1 = (unsigned int)(uintptr_t)&Bs1[row * LDK + seg * 16];

  v8f zero = {};
  v8f acc[4][2];
#pragma unroll
  for (int i = 0; i < 4; ++i)
#pragma unroll
    for (int j = 0; j < 2; ++j) acc[i][j] = zero;

  stage4(lds_a0, lds_b0, a_src, b_src);
  for (int k0 = 0; k0 < NI; k0 += 2 * BK) {
    stage4(lds_a1, lds_b1, a_src + k0 + BK, b_src + k0 + BK);
    wait_async4();
    __syncthreads();
    mma_tile(As0, Bs0, waveM, waveN, lane, acc);
    __syncthreads();
    if (k0 + 2 * BK < NI) {
      stage4(lds_a0, lds_b0, a_src + k0 + 2 * BK, b_src + k0 + 2 * BK);
      wait_async4();
    } else {
      wait_async0();
    }
    __syncthreads();
    mma_tile(As1, Bs1, waveM, waveN, lane, acc);
    __syncthreads();
  }

  const int ln = lane & 15;
  const int lh = (lane >> 4) << 3;
#pragma unroll
  for (int mi = 0; mi < 4; ++mi) {
#pragma unroll
    for (int ni = 0; ni < 2; ++ni) {
      int n = n0 + waveN * 32 + ni * 16 + ln;
      float bias = b2[e * NH + n];
#pragma unroll
      for (int j = 0; j < 8; ++j) {
        int m = mt * BM + waveM * 64 + mi * 16 + lh + j;
        if (m < cnt) {
          int orig = sorted[off + m];
          yffn[(size_t)orig * NH + n] = acc[mi][ni][j] + bias;
        }
      }
    }
  }
}

// ---------------- residual + LayerNorm ----------------
__global__ __launch_bounds__(256) void k_ln(
    const float* __restrict__ yffn, const float* __restrict__ x,
    const float* __restrict__ gamma, const float* __restrict__ beta,
    float* __restrict__ out) {
  __shared__ float s_sum[256];
  __shared__ float s_sq[256];
  const int t = blockIdx.x;
  const float* yr = yffn + (size_t)t * NH;
  const float* xr = x + (size_t)t * NH;
  float v[3];
  float s = 0.f, sq = 0.f;
#pragma unroll
  for (int i = 0; i < 3; ++i) {
    int c = threadIdx.x + i * 256;
    v[i] = yr[c] + xr[c];
    s += v[i];
    sq += v[i] * v[i];
  }
  s_sum[threadIdx.x] = s;
  s_sq[threadIdx.x] = sq;
  __syncthreads();
  for (int w = 128; w > 0; w >>= 1) {
    if (threadIdx.x < w) {
      s_sum[threadIdx.x] += s_sum[threadIdx.x + w];
      s_sq[threadIdx.x]  += s_sq[threadIdx.x + w];
    }
    __syncthreads();
  }
  float mu  = s_sum[0] * (1.0f / NH);
  float var = s_sq[0] * (1.0f / NH) - mu * mu;
  float inv = rsqrtf(var + LN_EPS);
#pragma unroll
  for (int i = 0; i < 3; ++i) {
    int c = threadIdx.x + i * 256;
    out[(size_t)t * NH + c] = (v[i] - mu) * inv * gamma[c] + beta[c];
  }
}

// ---------------- host-side launch ----------------
extern "C" void kernel_launch(void* const* d_in, const int* in_sizes, int n_in,
                              void* d_out, int out_size, void* d_ws, size_t ws_size,
                              hipStream_t stream) {
  const float* x   = (const float*)d_in[0];   // [1,T,H]
  const float* w1  = (const float*)d_in[1];   // [E,H,I]
  const float* b1  = (const float*)d_in[2];   // [E,1,I]
  const float* w2  = (const float*)d_in[3];   // [E,I,H]
  const float* b2  = (const float*)d_in[4];   // [E,1,H]
  const float* gam = (const float*)d_in[5];   // [H]
  const float* bet = (const float*)d_in[6];   // [H]
  const int*   eid = (const int*)d_in[7];     // [T]
  float* out = (float*)d_out;
  (void)in_sizes; (void)n_in; (void)out_size; (void)ws_size;

  char* wp = (char*)d_ws;
  auto carve = [&](size_t bytes) -> void* {
    void* p = (void*)wp;
    wp += (bytes + 255) & ~(size_t)255;
    return p;
  };
  int*    counts  = (int*)carve(NE * sizeof(int));
  int*    cursor  = (int*)carve(NE * sizeof(int));
  int*    offsets = (int*)carve(NE * sizeof(int));
  int*    sorted  = (int*)carve((size_t)NT * sizeof(int));
  bf16_t* xb      = (bf16_t*)carve((size_t)NT * NH * sizeof(bf16_t));
  bf16_t* w1t     = (bf16_t*)carve((size_t)NE * NH * NI * sizeof(bf16_t));
  bf16_t* w2t     = (bf16_t*)carve((size_t)NE * NI * NH * sizeof(bf16_t));
  bf16_t* inter   = (bf16_t*)carve((size_t)NT * NI * sizeof(bf16_t));
  float*  yffn    = (float*)carve((size_t)NT * NH * sizeof(float));

  // 1) precision conversion (+ weight transpose to [E][N][K]) into workspace
  k_f32_to_bf16<<<4096, 256, 0, stream>>>(x, xb, (size_t)NT * NH);
  k_transpose_bf16<<<dim3(NI / 32, NH / 32, NE), 256, 0, stream>>>(w1, w1t, NH, NI);
  k_transpose_bf16<<<dim3(NH / 32, NI / 32, NE), 256, 0, stream>>>(w2, w2t, NI, NH);

  // 2) token routing
  k_route_init<<<1, 32, 0, stream>>>(counts, cursor);
  k_route_count<<<NT / 256, 256, 0, stream>>>(eid, counts);
  k_route_scan<<<1, 1, 0, stream>>>(counts, offsets);
  k_route_fill<<<NT / 256, 256, 0, stream>>>(eid, offsets, cursor, sorted);

  // 3) grouped GEMM1 + GELU
  k_ffn1<<<dim3(NI / BN, NE * (NT / BM)), 256, 0, stream>>>(
      xb, w1t, b1, counts, offsets, sorted, inter);

  // 4) grouped GEMM2, scatter fp32 FFN output to token order
  k_ffn2<<<dim3(NH / BN, NE * (NT / BM)), 256, 0, stream>>>(
      inter, w2t, b2, counts, offsets, sorted, yffn);

  // 5) residual + LayerNorm
  k_ln<<<NT, 256, 0, stream>>>(yffn, x, gam, bet, out);
}
